// NeuralODE_51067161150121
// MI455X (gfx1250) — matrix operand order repaired and checked by reading, not compiled, not run on hardware
//
#include <hip/hip_runtime.h>
#include <math.h>

typedef __attribute__((ext_vector_type(16))) _Float16 v16h;
typedef __attribute__((ext_vector_type(8)))  float    v8f;

#if __has_builtin(__builtin_amdgcn_tanhf)
__device__ __forceinline__ float fast_tanh(float x) { return __builtin_amdgcn_tanhf(x); }
#else
__device__ __forceinline__ float fast_tanh(float x) { return tanhf(x); }
#endif

#define WMMA_F16(A, B, C) \
    __builtin_amdgcn_wmma_f32_16x16x32_f16(false, (A), false, (B), (short)0, (C), false, false)

// split an f32 into hi/lo f16 parts: x ~= (float)hi + (float)lo, |lo| <= ulp(hi)/2
__device__ __forceinline__ void f16split(float x, _Float16& hi, _Float16& lo) {
    _Float16 h = (_Float16)x;
    hi = h;
    lo = (_Float16)(x - (float)h);
}

__global__ void __launch_bounds__(32)
node_rk4_kernel(const float* __restrict__ state0,
                const float* __restrict__ W1, const float* __restrict__ b1,
                const float* __restrict__ W2, const float* __restrict__ b2,
                const float* __restrict__ W3, const float* __restrict__ b3,
                const float* __restrict__ log_scale,
                const int*   __restrict__ dt_i,
                float* __restrict__ out, int M)
{
    // ---- physical constants (folded in double, used in f32 like the reference) ----
    constexpr double MUd = 398600.4418, REd = 6378.137;
    constexpr double J2d = 0.00108262668, J3d = -2.53265648e-06;
    constexpr double J4d = -1.61962159e-06, J5d = -2.27296082e-07;
    const float MUf = (float)MUd;
    const float K2  = (float)(-1.5      * J2d * MUd * REd*REd);
    const float K3  = (float)(-2.5      * J3d * MUd * REd*REd*REd);
    const float K4  = (float)((15.0/8.0)* J4d * MUd * REd*REd*REd*REd);
    const float K5  = (float)(( 3.0/8.0)* J5d * MUd * REd*REd*REd*REd*REd);
    const float K5B = (float)((15.0/8.0)* J5d * MUd * REd*REd*REd*REd*REd);
    const float R_REF = 7000.0f, V_REF = 7.5f;
    const float A_REF = V_REF * V_REF / R_REF;

    const int  l   = threadIdx.x;
    const bool hic = (l >= 16);            // high K-group lanes
    const int  nlo = l & 15;
    const int  kb  = hic ? 16 : 0;

    __shared__ __align__(16) float hbuf[32];

    // ---- preload weights into WMMA B-operand layout, split into hi/lo f16 ----
    // B k order used here (contiguous, k=16h+s): VGPR r, lanes 0-15 hold (K=2r,2r+1 ; N=lane),
    //                        lanes 16-31 hold (K=16+2r,17+2r ; N=lane-16).
    // Per lane: B[j] = W[N_row][kb + j], N_row = nlo (+16 for second WMMA).
    v16h B1aH, B1aL, B1bH, B1bL, B2aH, B2aL, B2bH, B2bL;
    for (int j = 0; j < 16; ++j) {
        float w1a = 0.f, w1b = 0.f;
        if (!hic && j < 6) { w1a = W1[nlo * 6 + j]; w1b = W1[(nlo + 16) * 6 + j]; }
        _Float16 h, lo;
        f16split(w1a, h, lo);                          B1aH[j] = h; B1aL[j] = lo;
        f16split(w1b, h, lo);                          B1bH[j] = h; B1bL[j] = lo;
        f16split(W2[(nlo     ) * 32 + kb + j], h, lo); B2aH[j] = h; B2aL[j] = lo;
        f16split(W2[(nlo + 16) * 32 + kb + j], h, lo); B2bH[j] = h; B2bL[j] = lo;
    }
    const float b1v = b1[l];
    const float b2v = b2[l];
    const float w3r0 = W3[l], w3r1 = W3[32 + l], w3r2 = W3[64 + l];
    const float b30 = b3[0], b31 = b3[1], b32 = b3[2];
    const float escale = expf(log_scale[0]) * A_REF;
    const float dtf = (float)(dt_i[0]);

    float s0 = state0[0], s1 = state0[1], s2 = state0[2];
    float s3 = state0[3], s4 = state0[4], s5 = state0[5];

    const v8f acc0 = {0.f, 0.f, 0.f, 0.f, 0.f, 0.f, 0.f, 0.f};

    // Split-precision matvec: W*h ~= WH*hH + WH*hL + WL*hH (f32 accumulate,
    // dropped WL*hL term is O(2^-22)). The 3 WMMAs are independent (combine
    // element 0 with scalar adds) so they pipeline instead of chaining.
    auto dyn = [&](float px, float py, float pz, float vx, float vy, float vz,
                   float (&d)[6]) {
        const float ir = 1.f / R_REF, iv = 1.f / V_REF;
        const float snv[6] = {px * ir, py * ir, pz * ir, vx * iv, vy * iv, vz * iv};

        // ---- Layer 1: A = sn broadcast (K=0..5 live; high-K lanes hold zeros) ----
        v16h A1H, A1L;
        for (int j = 0; j < 16; ++j) { A1H[j] = (_Float16)0.f; A1L[j] = (_Float16)0.f; }
        for (int j = 0; j < 6; ++j) {
            float v = hic ? 0.f : snv[j];
            _Float16 h, lo;
            f16split(v, h, lo);
            A1H[j] = h; A1L[j] = lo;
        }
        v8f a0 = WMMA_F16(A1H, B1aH, acc0);
        v8f a1 = WMMA_F16(A1L, B1aH, acc0);
        v8f a2 = WMMA_F16(A1H, B1aL, acc0);
        v8f b0 = WMMA_F16(A1H, B1bH, acc0);
        v8f b1x = WMMA_F16(A1L, B1bH, acc0);
        v8f b2x = WMMA_F16(A1H, B1bL, acc0);
        float pre1 = hic ? (b0[0] + b1x[0] + b2x[0]) : (a0[0] + a1[0] + a2[0]);
        float h1 = fast_tanh(pre1 + b1v);

        // ---- redistribute h1 (one element per lane) into A-operand layout via LDS ----
        __syncthreads();
        hbuf[l] = h1;
        __syncthreads();
        const float4* hb4 = (const float4*)hbuf;
        const int base = hic ? 4 : 0;  // lanes<16: K=0..15 ; lanes>=16: K=16..31
        float4 q0 = hb4[base + 0];
        float4 q1 = hb4[base + 1];
        float4 q2 = hb4[base + 2];
        float4 q3 = hb4[base + 3];
        float f[16] = {q0.x, q0.y, q0.z, q0.w, q1.x, q1.y, q1.z, q1.w,
                       q2.x, q2.y, q2.z, q2.w, q3.x, q3.y, q3.z, q3.w};
        v16h A2H, A2L;
        for (int j = 0; j < 16; ++j) {
            _Float16 h, lo;
            f16split(f[j], h, lo);
            A2H[j] = h; A2L[j] = lo;
        }

        // ---- Layer 2 ----
        v8f c0 = WMMA_F16(A2H, B2aH, acc0);
        v8f c1 = WMMA_F16(A2L, B2aH, acc0);
        v8f c2 = WMMA_F16(A2H, B2aL, acc0);
        v8f e0 = WMMA_F16(A2H, B2bH, acc0);
        v8f e1 = WMMA_F16(A2L, B2bH, acc0);
        v8f e2 = WMMA_F16(A2H, B2bL, acc0);
        float pre2 = hic ? (e0[0] + e1[0] + e2[0]) : (c0[0] + c1[0] + c2[0]);
        float h2 = fast_tanh(pre2 + b2v);

        // ---- Layer 3 (3x32 matvec) as wave reduction ----
        float p0 = w3r0 * h2, p1 = w3r1 * h2, p2 = w3r2 * h2;
        for (int m = 16; m >= 1; m >>= 1) {
            p0 += __shfl_xor(p0, m, 32);
            p1 += __shfl_xor(p1, m, 32);
            p2 += __shfl_xor(p2, m, 32);
        }

        // ---- J2..J5 gravity (f32, wave-uniform; overlaps WMMA latency) ----
        float r2 = px * px + py * py + pz * pz;
        float r  = sqrtf(r2);
        float r3 = r2 * r, r5 = r3 * r2, r7 = r5 * r2, r9 = r7 * r2;
        float z2 = pz * pz, zr2 = z2 / r2, zr4 = zr2 * zr2;
        float ax = -MUf * px / r3, ay = -MUf * py / r3, az = -MUf * pz / r3;
        float c2g = K2 / r5;
        ax += c2g * px * (1.f - 5.f * zr2);
        ay += c2g * py * (1.f - 5.f * zr2);
        az += c2g * pz * (3.f - 5.f * zr2);
        float c3g = K3 / r7;
        float t3 = 3.f * pz - 7.f * pz * z2 / r2;
        ax += c3g * px * t3;
        ay += c3g * py * t3;
        az += c3g * (6.f * z2 - 7.f * z2 * zr2 - 0.6f * r2);
        float c4g = K4 / r7;
        float p4 = 1.f - 14.f * zr2 + 21.f * zr4;
        ax += c4g * px * p4;
        ay += c4g * py * p4;
        az += c4g * pz * (5.f - (70.f / 3.f) * zr2 + 21.f * zr4);
        float c5g = K5 / r9;
        float p5 = 35.f - 210.f * zr2 + 231.f * zr4;
        ax += c5g * px * pz * p5;
        ay += c5g * py * pz * p5;
        az += c5g * z2 * (105.f - 315.f * zr2 + 231.f * zr4) - K5B / r7;

        ax += (p0 + b30) * escale;
        ay += (p1 + b31) * escale;
        az += (p2 + b32) * escale;

        d[0] = vx; d[1] = vy; d[2] = vz; d[3] = ax; d[4] = ay; d[5] = az;
    };

    const float hdt = 0.5f * dtf;
    const float c6  = dtf / 6.f;
#pragma unroll 1
    for (int step = 0; step < M; ++step) {
        float k1[6], k2[6], k3[6], k4[6];
        dyn(s0, s1, s2, s3, s4, s5, k1);
        dyn(s0 + hdt * k1[0], s1 + hdt * k1[1], s2 + hdt * k1[2],
            s3 + hdt * k1[3], s4 + hdt * k1[4], s5 + hdt * k1[5], k2);
        dyn(s0 + hdt * k2[0], s1 + hdt * k2[1], s2 + hdt * k2[2],
            s3 + hdt * k2[3], s4 + hdt * k2[4], s5 + hdt * k2[5], k3);
        dyn(s0 + dtf * k3[0], s1 + dtf * k3[1], s2 + dtf * k3[2],
            s3 + dtf * k3[3], s4 + dtf * k3[4], s5 + dtf * k3[5], k4);
        s0 += c6 * (k1[0] + 2.f * (k2[0] + k3[0]) + k4[0]);
        s1 += c6 * (k1[1] + 2.f * (k2[1] + k3[1]) + k4[1]);
        s2 += c6 * (k1[2] + 2.f * (k2[2] + k3[2]) + k4[2]);
        s3 += c6 * (k1[3] + 2.f * (k2[3] + k3[3]) + k4[3]);
        s4 += c6 * (k1[4] + 2.f * (k2[4] + k3[4]) + k4[4]);
        s5 += c6 * (k1[5] + 2.f * (k2[5] + k3[5]) + k4[5]);
        if (l == 0) {
            out[step * 6 + 0] = s0;
            out[step * 6 + 1] = s1;
            out[step * 6 + 2] = s2;
            out[step * 6 + 3] = s3;
            out[step * 6 + 4] = s4;
            out[step * 6 + 5] = s5;
        }
    }
}

extern "C" void kernel_launch(void* const* d_in, const int* in_sizes, int n_in,
                              void* d_out, int out_size, void* d_ws, size_t ws_size,
                              hipStream_t stream) {
    // setup_inputs order:
    // 0:state0 1:eval_times 2:W1 3:b1 4:W2 5:b2 6:W3 7:b3 8:log_scale 9:t0 10:dt
    const float* state0 = (const float*)d_in[0];
    const float* W1 = (const float*)d_in[2];
    const float* b1 = (const float*)d_in[3];
    const float* W2 = (const float*)d_in[4];
    const float* b2 = (const float*)d_in[5];
    const float* W3 = (const float*)d_in[6];
    const float* b3 = (const float*)d_in[7];
    const float* ls = (const float*)d_in[8];
    const int*   dt = (const int*)d_in[10];
    const int M = in_sizes[1];   // number of eval times / RK4 steps

    node_rk4_kernel<<<1, 32, 0, stream>>>(state0, W1, b1, W2, b2, W3, b3, ls, dt,
                                          (float*)d_out, M);
}
